// EmbeddingMoE_2886218023430
// MI455X (gfx1250) — matrix operand order repaired and checked by reading, not compiled, not run on hardware
//
#include <hip/hip_runtime.h>
#include <hip/hip_bf16.h>
#include <cmath>

#define B_   4
#define SEQ_ 1024
#define D_   512
#define H_   2048
#define E_   4
#define CAP_ 512
#define V_   30522
#define T_   (B_*SEQ_)

typedef __attribute__((ext_vector_type(16))) __bf16 bf16x16;
typedef __attribute__((ext_vector_type(8)))  float  f32x8;

struct Q32 { uint4 a, b; };
__device__ __forceinline__ bf16x16 frag_cast(uint4 a, uint4 b) {
  Q32 q{a, b};
  return __builtin_bit_cast(bf16x16, q);
}

// CDNA5 async global->LDS copy (ASYNCcnt-tracked, bypasses VGPRs).
// lds_addr = low 32 bits of the generic __shared__ pointer (ISA: generic LDS
// address = {shared aperture, 32-bit LDS offset}).
__device__ __forceinline__ void async_load_b128(uint32_t lds_addr,
                                                const void* gptr) {
  asm volatile("global_load_async_to_lds_b128 %0, %1, off"
               :: "v"(lds_addr), "v"((unsigned long long)(size_t)gptr)
               : "memory");
}
#define WAIT_ASYNCCNT_4() asm volatile("s_wait_asynccnt 0x4" ::: "memory")
#define WAIT_ASYNCCNT_0() asm volatile("s_wait_asynccnt 0x0" ::: "memory")

// ---------------------------------------------------------------------------
// Embedding gather: h0[b,n,d] = emb_w[x[b,n], d]
// ---------------------------------------------------------------------------
__global__ __launch_bounds__(256) void embed_kernel(
    const int* __restrict__ x, const float* __restrict__ emb,
    float* __restrict__ h0) {
  int i = blockIdx.x * 256 + threadIdx.x;   // grid sized exactly T_*D_/256
  int tok = i >> 9;                          // D_ = 512
  int d   = i & (D_ - 1);
  h0[i] = emb[(size_t)x[tok] * D_ + d];
}

// ---------------------------------------------------------------------------
// Flat f32 -> bf16 convert
// ---------------------------------------------------------------------------
__global__ __launch_bounds__(256) void cvt_bf16_kernel(
    const float* __restrict__ src, __bf16* __restrict__ dst, int n) {
  int i = blockIdx.x * 256 + threadIdx.x;
  if (i < n) dst[i] = (__bf16)src[i];
}

// ---------------------------------------------------------------------------
// Tiled transpose + convert: src f32 [z][R][C] -> dst bf16 [z][C][R]
// block (32,8)
// ---------------------------------------------------------------------------
__global__ __launch_bounds__(256) void transpose_cvt_kernel(
    const float* __restrict__ src, __bf16* __restrict__ dst,
    int R, int C, size_t sBatch, size_t dBatch) {
  __shared__ float tile[32][33];
  src += (size_t)blockIdx.z * sBatch;
  dst += (size_t)blockIdx.z * dBatch;
  int c0 = blockIdx.x * 32, r0 = blockIdx.y * 32;
  int tx = threadIdx.x, ty = threadIdx.y;
#pragma unroll
  for (int i = 0; i < 4; ++i) {
    int r = r0 + ty + i * 8;
    tile[ty + i * 8][tx] = src[(size_t)r * C + c0 + tx];
  }
  __syncthreads();
#pragma unroll
  for (int i = 0; i < 4; ++i) {
    int c = c0 + ty + i * 8;
    dst[(size_t)c * R + r0 + tx] = (__bf16)tile[tx][ty + i * 8];
  }
}

// ---------------------------------------------------------------------------
// Top-2 gating for one MoE layer. One block per batch b.
// ---------------------------------------------------------------------------
__global__ __launch_bounds__(256) void gate_kernel(
    const float* __restrict__ h, const float* __restrict__ wg,
    int* __restrict__ oe1, int* __restrict__ op1, float* __restrict__ og1,
    int* __restrict__ oe2, int* __restrict__ op2, float* __restrict__ og2,
    int* __restrict__ slot_tok,        // [E_][B_][CAP_]
    float* __restrict__ auxSlot) {     // [B_] (this layer)
  __shared__ float sWg[D_ * E_];
  __shared__ int   sE1[SEQ_], sE2[SEQ_];
  __shared__ float sG1[SEQ_], sG2[SEQ_];
  __shared__ float sPart[256][E_];
  __shared__ float sSumRaw[E_];

  const int b = blockIdx.x;
  const int tid = threadIdx.x;

  for (int i = tid; i < D_ * E_; i += 256) sWg[i] = wg[i];
  for (int i = tid; i < E_ * CAP_; i += 256)
    slot_tok[(i / CAP_) * (B_ * CAP_) + b * CAP_ + (i % CAP_)] = -1;
  float part[E_] = {0.f, 0.f, 0.f, 0.f};
  __syncthreads();

  for (int n = tid; n < SEQ_; n += 256) {
    const float* xr = h + ((size_t)b * SEQ_ + n) * D_;
    float acc[E_] = {0.f, 0.f, 0.f, 0.f};
    for (int k = 0; k < D_; ++k) {
      float xv = xr[k];
      float4 w = *(const float4*)(sWg + k * 4);
      acc[0] += xv * w.x; acc[1] += xv * w.y;
      acc[2] += xv * w.z; acc[3] += xv * w.w;
    }
    float mx = fmaxf(fmaxf(acc[0], acc[1]), fmaxf(acc[2], acc[3]));
    float p[E_]; float s = 0.f;
#pragma unroll
    for (int e = 0; e < E_; ++e) { p[e] = expf(acc[e] - mx); s += p[e]; }
    float inv = 1.f / s;
#pragma unroll
    for (int e = 0; e < E_; ++e) { p[e] *= inv; part[e] += p[e]; }
    int i1 = 0; float b1 = p[0];
#pragma unroll
    for (int e = 1; e < E_; ++e) if (p[e] > b1) { b1 = p[e]; i1 = e; }
    float q[E_];
#pragma unroll
    for (int e = 0; e < E_; ++e) q[e] = p[e];
    q[i1] = -1.f;
    int i2 = 0; float b2 = q[0];
#pragma unroll
    for (int e = 1; e < E_; ++e) if (q[e] > b2) { b2 = q[e]; i2 = e; }
    float denom = b1 + b2 + 1e-9f;
    sE1[n] = i1; sE2[n] = i2;
    sG1[n] = b1 / denom; sG2[n] = b2 / denom;
  }
#pragma unroll
  for (int e = 0; e < E_; ++e) sPart[tid][e] = part[e];
  __syncthreads();
  for (int stride = 128; stride > 0; stride >>= 1) {
    if (tid < stride)
#pragma unroll
      for (int e = 0; e < E_; ++e) sPart[tid][e] += sPart[tid + stride][e];
    __syncthreads();
  }
  if (tid < E_) sSumRaw[tid] = sPart[0][tid];
  __syncthreads();

  if (tid == 0) {
    const size_t base = (size_t)b * SEQ_;
    int c1[E_] = {0, 0, 0, 0};
    for (int n = 0; n < SEQ_; ++n) {
      int e = sE1[n]; int pos = c1[e]++;
      oe1[base + n] = e;
      if (pos < CAP_) {
        op1[base + n] = pos; og1[base + n] = sG1[n];
        slot_tok[(e * B_ + b) * CAP_ + pos] = n;
      } else { op1[base + n] = 0; og1[base + n] = 0.f; }
    }
    float aux = 0.f;
#pragma unroll
    for (int e = 0; e < E_; ++e)
      aux += (sSumRaw[e] * (1.f / SEQ_)) * ((float)c1[e] * (1.f / SEQ_));
    auxSlot[b] = aux;
    int c2[E_];
#pragma unroll
    for (int e = 0; e < E_; ++e) c2[e] = c1[e] < CAP_ ? c1[e] : CAP_;
    for (int n = 0; n < SEQ_; ++n) {
      int e = sE2[n]; int pos = c2[e]++;
      oe2[base + n] = e;
      if (pos < CAP_) {
        op2[base + n] = pos; og2[base + n] = sG2[n];
        slot_tok[(e * B_ + b) * CAP_ + pos] = n;
      } else { op2[base + n] = 0; og2[base + n] = 0.f; }
    }
  }
}

// ---------------------------------------------------------------------------
// Gather tokens into expert slots (bf16). One block per [e][b][c] slot row.
// ---------------------------------------------------------------------------
__global__ __launch_bounds__(256) void gather_kernel(
    const float* __restrict__ h, const int* __restrict__ slot_tok,
    __bf16* __restrict__ ein) {
  int row = blockIdx.x;                 // e*B_*CAP_ + b*CAP_ + c
  int tok = slot_tok[row];
  int b = (row / CAP_) % B_;
  __bf16* dst = ein + (size_t)row * D_;
  if (tok < 0) {
    for (int d = threadIdx.x; d < D_; d += 256) dst[d] = (__bf16)0.f;
  } else {
    const float* src = h + ((size_t)b * SEQ_ + tok) * D_;
    for (int d = threadIdx.x; d < D_; d += 256) dst[d] = (__bf16)src[d];
  }
}

// ---------------------------------------------------------------------------
// Combine expert outputs back to tokens. One block per token.
// ---------------------------------------------------------------------------
__global__ __launch_bounds__(256) void combine_kernel(
    const float* __restrict__ eo,
    const int* __restrict__ ie1, const int* __restrict__ ip1, const float* __restrict__ ig1,
    const int* __restrict__ ie2, const int* __restrict__ ip2, const float* __restrict__ ig2,
    float* __restrict__ hout, __bf16* __restrict__ hbf) {
  int t = blockIdx.x;                    // b*SEQ_ + n
  int b = t / SEQ_;
  int E1 = ie1[t], P1 = ip1[t], E2 = ie2[t], P2 = ip2[t];
  float G1 = ig1[t], G2 = ig2[t];
  const float* r1 = eo + (((size_t)E1 * B_ + b) * CAP_ + P1) * D_;
  const float* r2 = eo + (((size_t)E2 * B_ + b) * CAP_ + P2) * D_;
  for (int d = threadIdx.x; d < D_; d += 256) {
    float v = G1 * r1[d] + G2 * r2[d];
    hout[(size_t)t * D_ + d] = v;
    if (hbf) hbf[(size_t)t * D_ + d] = (__bf16)v;
  }
}

// ---------------------------------------------------------------------------
// bf16 WMMA GEMM: C[M][N] = A[M][K] * B[N][K]^T (+bias) (gelu) (bf16 out)
// 128x128x32 tile per 256-thread block; 8 waves, each 64x32 via 8 WMMA accs.
// Double-buffered LDS filled by GLOBAL_LOAD_ASYNC_TO_LDS_B128: stage k+1
// streams in (ASYNCcnt) while stage k runs WMMAs. 80 B padded rows keep the
// fragment ds_load_b128s 16 B aligned; lane half h reads contiguous k = 16h..16h+15 for A and B alike.
// ---------------------------------------------------------------------------
template <bool OUT_BF16, bool GELU_ACT, bool ADD_BIAS>
__global__ __launch_bounds__(256) void gemm_bf16_kernel(
    const __bf16* __restrict__ A, const __bf16* __restrict__ B,
    void* __restrict__ Cout, const float* __restrict__ bias,
    int M, int N, int K,
    size_t aBatch, size_t bBatch, size_t cBatch, int bDiv) {
  constexpr int LDB = 80;                  // LDS row pitch in bytes
  __shared__ __bf16 sA[2][128 * (LDB / 2)];
  __shared__ __bf16 sB[2][128 * (LDB / 2)];

  const int tid = threadIdx.x;
  const int z = blockIdx.z;
  A += (size_t)z * aBatch;
  B += (size_t)(z / bDiv) * bBatch;

  const int N0 = blockIdx.x * 128;
  const int M0 = blockIdx.y * 128;
  const int lane = tid & 31;
  const int wid = tid >> 5;
  const int wm = (wid & 1) * 64;           // wave M offset within tile
  const int wn = (wid >> 1) * 32;          // wave N offset within tile
  const int lh = lane >> 4;                // lane half (0/1)
  const int lr = lane & 15;

  // Per-thread staging chunks: c = tid + s*256, row = c>>2, 16B sub = c&3.
  const int row0 = tid >> 2, sub0 = tid & 3;
  const int row1 = (tid + 256) >> 2, sub1 = (tid + 256) & 3;
  const uint32_t ldsA0 = (uint32_t)(size_t)&sA[0][0];
  const uint32_t ldsB0 = (uint32_t)(size_t)&sB[0][0];
  const uint32_t ldsA1 = (uint32_t)(size_t)&sA[1][0];
  const uint32_t ldsB1 = (uint32_t)(size_t)&sB[1][0];

  int gnr0 = N0 + row0; if (gnr0 >= N) gnr0 = N - 1;   // clamp: cols >= N never stored
  int gnr1 = N0 + row1; if (gnr1 >= N) gnr1 = N - 1;

  auto issue_stage = [&](int kt, int buf) {
    uint32_t la = buf ? ldsA1 : ldsA0;
    uint32_t lb = buf ? ldsB1 : ldsB0;
    async_load_b128(la + row0 * LDB + sub0 * 16,
                    A + (size_t)(M0 + row0) * K + kt + sub0 * 8);
    async_load_b128(lb + row0 * LDB + sub0 * 16,
                    B + (size_t)gnr0 * K + kt + sub0 * 8);
    async_load_b128(la + row1 * LDB + sub1 * 16,
                    A + (size_t)(M0 + row1) * K + kt + sub1 * 8);
    async_load_b128(lb + row1 * LDB + sub1 * 16,
                    B + (size_t)gnr1 * K + kt + sub1 * 8);
  };

  f32x8 acc[4][2];
  const f32x8 zero = {0.f, 0.f, 0.f, 0.f, 0.f, 0.f, 0.f, 0.f};
#pragma unroll
  for (int i = 0; i < 4; ++i)
#pragma unroll
    for (int j = 0; j < 2; ++j) acc[i][j] = zero;

  issue_stage(0, 0);                        // prologue: stage 0 in flight

  for (int kt = 0, it = 0; kt < K; kt += 32, ++it) {
    const int buf = it & 1;
    if (kt + 32 < K) {
      issue_stage(kt + 32, buf ^ 1);        // overlap next stage with compute
      WAIT_ASYNCCNT_4();                    // in-order: current stage landed
    } else {
      WAIT_ASYNCCNT_0();
    }
    __syncthreads();                        // all waves' tiles visible

    const char* baseA = (const char*)&sA[buf][0];
    const char* baseB = (const char*)&sB[buf][0];
    bf16x16 bfrag[2];
#pragma unroll
    for (int j = 0; j < 2; ++j) {
      const char* bp = baseB + (wn + j * 16 + lr) * LDB + lh * 32;
      bfrag[j] = frag_cast(*(const uint4*)bp, *(const uint4*)(bp + 16));
    }
#pragma unroll
    for (int i = 0; i < 4; ++i) {
      const char* ap = baseA + (wm + i * 16 + lr) * LDB + lh * 32;
      bf16x16 af = frag_cast(*(const uint4*)ap, *(const uint4*)(ap + 16));
#pragma unroll
      for (int j = 0; j < 2; ++j)
        acc[i][j] = __builtin_amdgcn_wmma_f32_16x16x32_bf16(
            false, af, false, bfrag[j], (short)0, acc[i][j], false, false);
    }
    __syncthreads();                        // reads done before buf is rewritten
  }

  // Epilogue: C/D layout -> VGPR r, lanes 0-15: M=r, lanes 16-31: M=r+8; N=lr.
#pragma unroll
  for (int i = 0; i < 4; ++i) {
#pragma unroll
    for (int j = 0; j < 2; ++j) {
      int n = N0 + wn + j * 16 + lr;
      if (n < N) {
#pragma unroll
        for (int r = 0; r < 8; ++r) {
          int m = M0 + wm + i * 16 + lh * 8 + r;
          float v = acc[i][j][r];
          if (ADD_BIAS) v += bias[n];
          if (GELU_ACT) v = 0.5f * v * (1.f + erff(v * 0.70710678118654752f));
          size_t idx = (size_t)z * cBatch + (size_t)m * N + n;
          if (OUT_BF16) ((__bf16*)Cout)[idx] = (__bf16)v;
          else          ((float*)Cout)[idx] = v;
        }
      }
    }
  }
}

// ---------------------------------------------------------------------------
// Aux finalize: deterministic fixed-order sum of 8 partials -> last out elem.
// ---------------------------------------------------------------------------
__global__ void finalize_kernel(const float* __restrict__ auxParts,
                                float* __restrict__ out, size_t idx) {
  float s = 0.f;
#pragma unroll
  for (int i = 0; i < 2 * B_; ++i) s += auxParts[i];
  out[idx] = s;
}

// ---------------------------------------------------------------------------
extern "C" void kernel_launch(void* const* d_in, const int* in_sizes, int n_in,
                              void* d_out, int out_size, void* d_ws, size_t ws_size,
                              hipStream_t stream) {
  const int*   x     = (const int*)d_in[0];
  const float* emb_w = (const float*)d_in[1];
  const float* wg1   = (const float*)d_in[2];
  const float* w1a   = (const float*)d_in[3];
  const float* w2a   = (const float*)d_in[4];
  const float* wg2   = (const float*)d_in[5];
  const float* w1b   = (const float*)d_in[6];
  const float* w2b   = (const float*)d_in[7];
  const float* mlm_w = (const float*)d_in[8];
  const float* mlm_b = (const float*)d_in[9];
  float* out = (float*)d_out;

  char* ws = (char*)d_ws;
  size_t off = 0;
  auto alloc = [&](size_t bytes) -> char* {
    char* p = ws + off;
    off += (bytes + 255) & ~(size_t)255;
    return p;
  };
  float*  h0    = (float*)alloc((size_t)T_ * D_ * 4);          // reused for h2 f32
  float*  h1    = (float*)alloc((size_t)T_ * D_ * 4);
  __bf16* h2bf  = (__bf16*)alloc((size_t)T_ * D_ * 2);
  __bf16* ein   = (__bf16*)alloc((size_t)E_ * B_ * CAP_ * D_ * 2);
  __bf16* Hbuf  = (__bf16*)alloc((size_t)E_ * B_ * CAP_ * H_ * 2);
  float*  eout  = (float*)alloc((size_t)E_ * B_ * CAP_ * D_ * 4);
  __bf16* w1at  = (__bf16*)alloc((size_t)E_ * H_ * D_ * 2);
  __bf16* w2at  = (__bf16*)alloc((size_t)E_ * D_ * H_ * 2);
  __bf16* w1bt  = (__bf16*)alloc((size_t)E_ * H_ * D_ * 2);
  __bf16* w2bt  = (__bf16*)alloc((size_t)E_ * D_ * H_ * 2);
  __bf16* mlmwb = (__bf16*)alloc((size_t)V_ * D_ * 2);
  int*   ge1[2]; int* gp1[2]; float* gg1[2];
  int*   ge2[2]; int* gp2[2]; float* gg2[2];
  int*   slot[2];
  for (int l = 0; l < 2; ++l) {
    ge1[l] = (int*)alloc(T_ * 4);   gp1[l] = (int*)alloc(T_ * 4);
    gg1[l] = (float*)alloc(T_ * 4);
    ge2[l] = (int*)alloc(T_ * 4);   gp2[l] = (int*)alloc(T_ * 4);
    gg2[l] = (float*)alloc(T_ * 4);
    slot[l] = (int*)alloc(E_ * B_ * CAP_ * 4);
  }
  float* auxParts = (float*)alloc(2 * B_ * 4);

  // 1. Embedding
  embed_kernel<<<(T_ * D_) / 256, 256, 0, stream>>>(x, emb_w, h0);

  // 2. Weight conversions (bf16, [N][K] layout)
  {
    int n = V_ * D_;
    cvt_bf16_kernel<<<(n + 255) / 256, 256, 0, stream>>>(mlm_w, mlmwb, n);
    dim3 blk(32, 8);
    transpose_cvt_kernel<<<dim3(H_ / 32, D_ / 32, E_), blk, 0, stream>>>(
        w1a, w1at, D_, H_, (size_t)D_ * H_, (size_t)D_ * H_);
    transpose_cvt_kernel<<<dim3(H_ / 32, D_ / 32, E_), blk, 0, stream>>>(
        w1b, w1bt, D_, H_, (size_t)D_ * H_, (size_t)D_ * H_);
    transpose_cvt_kernel<<<dim3(D_ / 32, H_ / 32, E_), blk, 0, stream>>>(
        w2a, w2at, H_, D_, (size_t)H_ * D_, (size_t)H_ * D_);
    transpose_cvt_kernel<<<dim3(D_ / 32, H_ / 32, E_), blk, 0, stream>>>(
        w2b, w2bt, H_, D_, (size_t)H_ * D_, (size_t)H_ * D_);
  }

  // 3. MoE layers
  for (int l = 0; l < 2; ++l) {
    const float* hin = (l == 0) ? h0 : h1;
    const float* wg  = (l == 0) ? wg1 : wg2;
    __bf16* w1t = (l == 0) ? w1at : w1bt;
    __bf16* w2t = (l == 0) ? w2at : w2bt;
    float*  hout = (l == 0) ? h1 : h0;
    __bf16* hbf  = (l == 0) ? (__bf16*)nullptr : h2bf;

    gate_kernel<<<B_, 256, 0, stream>>>(hin, wg,
        ge1[l], gp1[l], gg1[l], ge2[l], gp2[l], gg2[l],
        slot[l], auxParts + l * B_);
    gather_kernel<<<E_ * B_ * CAP_, 256, 0, stream>>>(hin, slot[l], ein);
    gemm_bf16_kernel<true, true, false>
        <<<dim3(H_ / 128, CAP_ / 128, E_ * B_), 256, 0, stream>>>(
            ein, w1t, Hbuf, nullptr, CAP_, H_, D_,
            (size_t)CAP_ * D_, (size_t)H_ * D_, (size_t)CAP_ * H_, B_);
    gemm_bf16_kernel<false, false, false>
        <<<dim3(D_ / 128, CAP_ / 128, E_ * B_), 256, 0, stream>>>(
            Hbuf, w2t, eout, nullptr, CAP_, D_, H_,
            (size_t)CAP_ * H_, (size_t)D_ * H_, (size_t)CAP_ * D_, B_);
    combine_kernel<<<T_, 256, 0, stream>>>(eout,
        ge1[l], gp1[l], gg1[l], ge2[l], gp2[l], gg2[l], hout, hbf);
  }

  // 4. MLM head: logits = h2bf x mlm_w^T + mlm_b   (M=4096, N=30522, K=512)
  gemm_bf16_kernel<false, false, true>
      <<<dim3((V_ + 127) / 128, T_ / 128, 1), 256, 0, stream>>>(
          h2bf, mlmwb, out, mlm_b, T_, V_, D_, 0, 0, 0, 1);

  // 5. Aux scalar (last output element)
  finalize_kernel<<<1, 1, 0, stream>>>(auxParts, out, (size_t)out_size - 1);
}
